// CausalMultiHeadSelfAttention_61100204753081
// MI455X (gfx1250) — matrix operand order repaired and checked by reading, not compiled, not run on hardware
//
#include <hip/hip_runtime.h>
#include <hip/hip_bf16.h>

// ---------------------------------------------------------------------------
// CausalMultiHeadSelfAttention for MI455X (gfx1250), wave32 + WMMA f16 + TDM.
// B=4, S=2048, D=1024, H=16, dk=64, THETA=10000.
// ---------------------------------------------------------------------------

typedef __attribute__((ext_vector_type(16))) _Float16 v16h;
typedef __attribute__((ext_vector_type(2)))  _Float16 v2h;
typedef __attribute__((ext_vector_type(8)))  float    v8f;
typedef __attribute__((ext_vector_type(4)))  unsigned int u32x4;
typedef __attribute__((ext_vector_type(8)))  int      i32x8;
typedef __attribute__((ext_vector_type(4)))  int      i32x4;

#define BATCH 4
#define SEQ   2048
#define DIM   1024
#define HEADS 16
#define DK    64

#if defined(__has_builtin)
#if __has_builtin(__builtin_amdgcn_tensor_load_to_lds) && \
    __has_builtin(__builtin_amdgcn_s_wait_tensorcnt)
#define HAVE_TDM 1
#endif
#endif

static __device__ __forceinline__ v8f wmma_f16(v16h a, v16h b, v8f c) {
  return __builtin_amdgcn_wmma_f32_16x16x32_f16(
      /*neg_a=*/false, a, /*neg_b=*/false, b,
      /*c_mod=*/(short)0, c, /*reuse_a=*/false, /*reuse_b=*/false);
}

// A fragment (16x32 f16): lane holds row M = row0 + (lane&15).
// ISA layout: v=0..3 -> K = 2v + 8*(lane>=16); v=4..7 -> K = 16 + 2(v-4) + 8*(lane>=16).
static __device__ __forceinline__ v16h frag_a(const _Float16* base, int row0,
                                              int k0, int ld) {
  const int lane = threadIdx.x & 31;
  const _Float16* p = base + (row0 + (lane & 15)) * ld;
  const int hs = (lane >> 4) * 8;
  v16h a;
#pragma unroll
  for (int v = 0; v < 4; ++v) {
    v2h t = *(const v2h*)(p + k0 + hs + 2 * v);
    a[2 * v] = t.x; a[2 * v + 1] = t.y;
  }
#pragma unroll
  for (int v = 4; v < 8; ++v) {
    v2h t = *(const v2h*)(p + k0 + 16 + hs + 2 * (v - 4));
    a[2 * v] = t.x; a[2 * v + 1] = t.y;
  }
  return a;
}

// B fragment (32x16 f16): lane holds col N = col0 + (lane&15).
// Layout: v=0..3 -> K = 2v + 8*(lane>=16); v=4..7 -> K = 16 + 2(v-4) + 8*(lane>=16). Tile stored [N][K], K contiguous.
static __device__ __forceinline__ v16h frag_b(const _Float16* base, int col0,
                                              int k0, int ld) {
  const int lane = threadIdx.x & 31;
  const _Float16* p = base + (col0 + (lane & 15)) * ld;
  const int hs = (lane >> 4) * 8;
  v16h b;
#pragma unroll
  for (int v = 0; v < 4; ++v) {
    v2h t = *(const v2h*)(p + k0 + hs + 2 * v);
    b[2 * v] = t.x; b[2 * v + 1] = t.y;
  }
#pragma unroll
  for (int v = 4; v < 8; ++v) {
    v2h t = *(const v2h*)(p + k0 + 16 + hs + 2 * (v - 4));
    b[2 * v] = t.x; b[2 * v + 1] = t.y;
  }
  return b;
}

// ---------------------------------------------------------------------------
// Kernel 1: qkv = x @ Wqkv^T, fused RoPE, scatter to head-major f16 Q/K/V.
// Grid: (3072/128, 8192/128), block 256 = 8 waves, wave tile 32x64 (8 WMMAs).
// ---------------------------------------------------------------------------
__global__ __launch_bounds__(256) void qkv_rope_kernel(
    const float* __restrict__ x, const float* __restrict__ Wqkv,
    _Float16* __restrict__ Qb, _Float16* __restrict__ Kb,
    _Float16* __restrict__ Vb) {
  __shared__ _Float16 As[128 * 36];  // [m][k]
  __shared__ _Float16 Bs[128 * 36];  // [n][k]

  const int tid = threadIdx.x;
  const int lane = tid & 31;
  const int w = tid >> 5;
  const int wm = w >> 1, wn = w & 1;
  const int m0 = blockIdx.y * 128;   // row in flattened (b,s)
  const int n0 = blockIdx.x * 128;   // col in o (0..3071)

  v8f acc[2][4] = {};

  for (int kb = 0; kb < DIM; kb += 32) {
    __syncthreads();
    for (int i = tid; i < 128 * 32; i += 256) {
      int r = i >> 5, c = i & 31;
      As[r * 36 + c] = (_Float16)x[(m0 + r) * DIM + kb + c];
    }
    for (int i = tid; i < 128 * 32; i += 256) {
      int r = i >> 5, c = i & 31;
      Bs[r * 36 + c] = (_Float16)Wqkv[(n0 + r) * DIM + kb + c];
    }
    __syncthreads();
    v16h a0 = frag_a(As, wm * 32, 0, 36);
    v16h a1 = frag_a(As, wm * 32 + 16, 0, 36);
#pragma unroll
    for (int tn = 0; tn < 4; ++tn) {
      v16h b = frag_b(Bs, wn * 64 + tn * 16, 0, 36);
      acc[0][tn] = wmma_f16(a0, b, acc[0][tn]);
      acc[1][tn] = wmma_f16(a1, b, acc[1][tn]);
    }
  }

  // Epilogue: each wave's 64-wide N slice is exactly one head of q, k or v.
  const int region = n0 >> 10;                    // 0=q, 1=k, 2=v
  const int h = ((n0 & 1023) >> 6) + wn;          // head index
  _Float16* dst = (region == 0) ? Qb : ((region == 1) ? Kb : Vb);

#pragma unroll
  for (int tm = 0; tm < 2; ++tm) {
#pragma unroll
    for (int tn = 0; tn < 4; ++tn) {
      const int col = tn * 16 + (lane & 15);      // j in [0,64) within head
      v8f c = acc[tm][tn];
      if (region < 2) {
        const int i2 = col & ~1;                  // pair base index (even)
        const float freq = expf(-(float)i2 * (logf(10000.0f) / (float)DK));
#pragma unroll
        for (int r = 0; r < 8; ++r) {
          int m = m0 + wm * 32 + tm * 16 + r + 8 * (lane >> 4);
          int s = m & (SEQ - 1);
          int bidx = m >> 11;
          float ang = (float)s * freq;
          float cs = cosf(ang), sn = sinf(ang);
          float val = c[r];
          float part = __shfl_xor(val, 1, 32);    // pair partner = lane^1
          float o = (col & 1) ? (sn * part + cs * val)
                              : (cs * val - sn * part);
          dst[(((size_t)h * BATCH + bidx) * SEQ + s) * DK + col] = (_Float16)o;
        }
      } else {
#pragma unroll
        for (int r = 0; r < 8; ++r) {
          int m = m0 + wm * 32 + tm * 16 + r + 8 * (lane >> 4);
          int s = m & (SEQ - 1);
          int bidx = m >> 11;
          dst[(((size_t)h * BATCH + bidx) * SEQ + s) * DK + col] =
              (_Float16)c[r];
        }
      }
    }
  }
}

// ---------------------------------------------------------------------------
// Kernel 2: causal flash attention. Grid (S/64, H*B), 128 threads = 4 waves.
// K tile staged by the Tensor Data Mover (with LDS row padding); V staged
// transposed manually; each wave owns a 16-row q strip (wave-local softmax).
// ---------------------------------------------------------------------------
__global__ __launch_bounds__(128) void attn_kernel(
    const _Float16* __restrict__ Qb, const _Float16* __restrict__ Kb,
    const _Float16* __restrict__ Vb, _Float16* __restrict__ AO) {
  __shared__ _Float16 Qs[64 * 72];      // [q][dk]
  __shared__ _Float16 Ks[64 * 72];      // [key][dk]   (TDM destination)
  __shared__ _Float16 Vt[64 * 72];      // [dk][key]   (transposed)
  __shared__ _Float16 Ps[4][16 * 72];   // per-wave P strip [m][key]

  const int tid = threadIdx.x;
  const int lane = tid & 31;
  const int w = tid >> 5;
  const int hb = blockIdx.y;            // h*BATCH + b
  const int q0 = blockIdx.x * 64;
  const size_t slab = (size_t)hb * SEQ * DK;
  const float scale = 0.125f;           // 1/sqrt(64)

  {  // stage Q tile (f16, u32 copies)
    const uint32_t* Qg32 = (const uint32_t*)(Qb + slab + (size_t)q0 * DK);
    uint32_t* Qs32 = (uint32_t*)Qs;
    for (int i = tid; i < 64 * 32; i += 128) {
      int r = i >> 5, c = i & 31;
      Qs32[r * 36 + c] = Qg32[r * 32 + c];
    }
  }
  __syncthreads();
  v16h aq0 = frag_a(Qs, w * 16, 0, 72);
  v16h aq1 = frag_a(Qs, w * 16, 32, 72);

  v8f oacc[4] = {};
  float mrun[8], lrun[8];
#pragma unroll
  for (int r = 0; r < 8; ++r) { mrun[r] = -__builtin_inff(); lrun[r] = 0.0f; }

  const int kbmax = q0 >> 6;
  for (int kb = 0; kb <= kbmax; ++kb) {
    __syncthreads();  // previous Ks/Vt consumers are done

#ifdef HAVE_TDM
    if (w == 0) {
      // TDM: 64x64 2-byte tile, row stride 64 elems in memory, row-padded in
      // LDS: pad_interval = 32 DWORDs (one 128B row), pad_amount = 4 DWORDs
      // (16B) -> LDS row stride 72 halfs, matching frag loads.
      unsigned long long ga =
          (unsigned long long)(uintptr_t)(Kb + slab + (size_t)(kb * 64) * DK);
      unsigned int lds = (unsigned int)(uintptr_t)(&Ks[0]);
      u32x4 g0;
      g0[0] = 1u;                                      // count=1 (user D#)
      g0[1] = lds;                                     // lds_addr
      g0[2] = (unsigned int)ga;                        // global_addr[31:0]
      g0[3] = (unsigned int)(ga >> 32) | (2u << 30);   // ga[56:32] | type=2
      i32x8 g1;
      g1[0] = (1 << 16) | (1 << 20) | (4 << 22) | (3 << 25);
      //       data_size=2B | pad_en | pad_interval=32dw | pad_amount=4dw
      g1[1] = (64 << 16);        // tensor_dim0[15:0] = 64
      g1[2] = (64 << 16);        // tensor_dim0 hi=0 | tensor_dim1 lo=64
      g1[3] = (64 << 16);        // tensor_dim1 hi=0 | tile_dim0=64
      g1[4] = 64;                // tile_dim1=64 | tile_dim2=0
      g1[5] = 64;                // tensor_dim0_stride lo = 64
      g1[6] = 0;                 // stride0 hi | stride1 lo
      g1[7] = 0;
      i32x4 z4 = {0, 0, 0, 0};   // groups 2/3 unused (2D tensor)
      i32x8 z8 = {0, 0, 0, 0, 0, 0, 0, 0};
      __builtin_amdgcn_tensor_load_to_lds(g0, g1, z4, z4, z8, 0);
    }
#else
    {  // fallback: manual K tile copy [key][dk]
      const uint32_t* Kg32 =
          (const uint32_t*)(Kb + slab + (size_t)(kb * 64) * DK);
      uint32_t* Ks32 = (uint32_t*)Ks;
      for (int i = tid; i < 64 * 32; i += 128) {
        int r = i >> 5, c = i & 31;
        Ks32[r * 36 + c] = Kg32[r * 32 + c];
      }
    }
#endif
    {  // stage V transposed [dk][key]
      const _Float16* Vg = Vb + slab + (size_t)(kb * 64) * DK;
      for (int i = tid; i < 64 * 64; i += 128) {
        int key = i >> 6, n = i & 63;
        Vt[n * 72 + key] = Vg[key * DK + n];
      }
    }
#ifdef HAVE_TDM
    if (w == 0) __builtin_amdgcn_s_wait_tensorcnt(0);
#endif
    __syncthreads();

    // scores: 16x64 strip = 4 tiles x 2 k-steps
    v8f sc[4];
#pragma unroll
    for (int tn = 0; tn < 4; ++tn) {
      v8f s = {};
      s = wmma_f16(aq0, frag_b(Ks, tn * 16, 0, 72), s);
      s = wmma_f16(aq1, frag_b(Ks, tn * 16, 32, 72), s);
      sc[tn] = s;
    }

    const bool diag = (kb == kbmax);
#pragma unroll
    for (int tn = 0; tn < 4; ++tn) {
#pragma unroll
      for (int r = 0; r < 8; ++r) {
        float v = sc[tn][r] * scale;
        if (diag) {
          int mg = q0 + w * 16 + r + 8 * (lane >> 4);
          int ng = kb * 64 + tn * 16 + (lane & 15);
          if (ng > mg) v = -__builtin_inff();
        }
        sc[tn][r] = v;
      }
    }

    // online softmax (row-wise; a C-row spans one 16-lane half)
#pragma unroll
    for (int r = 0; r < 8; ++r) {
      float t = fmaxf(fmaxf(sc[0][r], sc[1][r]), fmaxf(sc[2][r], sc[3][r]));
      t = fmaxf(t, __shfl_xor(t, 1, 32));
      t = fmaxf(t, __shfl_xor(t, 2, 32));
      t = fmaxf(t, __shfl_xor(t, 4, 32));
      t = fmaxf(t, __shfl_xor(t, 8, 32));
      float mn = fmaxf(mrun[r], t);
      float corr = __expf(mrun[r] - mn);
      float rs = 0.0f;
#pragma unroll
      for (int tn = 0; tn < 4; ++tn) {
        float p = __expf(sc[tn][r] - mn);
        sc[tn][r] = p;
        rs += p;
      }
      rs += __shfl_xor(rs, 1, 32);
      rs += __shfl_xor(rs, 2, 32);
      rs += __shfl_xor(rs, 4, 32);
      rs += __shfl_xor(rs, 8, 32);
      lrun[r] = lrun[r] * corr + rs;
      mrun[r] = mn;
#pragma unroll
      for (int to = 0; to < 4; ++to) oacc[to][r] *= corr;
    }

    // C-layout -> A-layout reshape of P through per-wave LDS strip
    _Float16* myP = Ps[w];
#pragma unroll
    for (int tn = 0; tn < 4; ++tn) {
#pragma unroll
      for (int r = 0; r < 8; ++r) {
        int row = r + 8 * (lane >> 4);
        int colp = tn * 16 + (lane & 15);
        myP[row * 72 + colp] = (_Float16)sc[tn][r];
      }
    }
    __threadfence_block();  // order LDS store -> ds_load within wave

    v16h ap0 = frag_a(myP, 0, 0, 72);
    v16h ap1 = frag_a(myP, 0, 32, 72);
#pragma unroll
    for (int to = 0; to < 4; ++to) {
      oacc[to] = wmma_f16(ap0, frag_b(Vt, to * 16, 0, 72), oacc[to]);
      oacc[to] = wmma_f16(ap1, frag_b(Vt, to * 16, 32, 72), oacc[to]);
    }
  }

  // normalize and write f16 attention output [b][s][h*64+j]
  const int h = hb >> 2, b = hb & 3;
#pragma unroll
  for (int to = 0; to < 4; ++to) {
#pragma unroll
    for (int r = 0; r < 8; ++r) {
      int row = q0 + w * 16 + r + 8 * (lane >> 4);
      int col = h * DK + to * 16 + (lane & 15);
      float v = oacc[to][r] / lrun[r];
      AO[((size_t)b * SEQ + row) * DIM + col] = (_Float16)v;
    }
  }
}

// ---------------------------------------------------------------------------
// Kernel 3: out = AO @ Wout^T (fp32 output, 'bso' layout).
// Grid: (1024/128, 8192/128), block 256 = 8 waves, wave tile 32x64.
// ---------------------------------------------------------------------------
__global__ __launch_bounds__(256) void outproj_kernel(
    const _Float16* __restrict__ AO, const float* __restrict__ Wout,
    float* __restrict__ out) {
  __shared__ _Float16 As[128 * 36];
  __shared__ _Float16 Bs[128 * 36];

  const int tid = threadIdx.x;
  const int lane = tid & 31;
  const int w = tid >> 5;
  const int wm = w >> 1, wn = w & 1;
  const int m0 = blockIdx.y * 128;
  const int n0 = blockIdx.x * 128;

  v8f acc[2][4] = {};

  for (int kb = 0; kb < DIM; kb += 32) {
    __syncthreads();
    {  // A already f16: u32 copies
      const uint32_t* Ag32 = (const uint32_t*)AO;
      uint32_t* As32 = (uint32_t*)As;
      const int kb32 = kb >> 1;
      for (int i = tid; i < 128 * 16; i += 256) {
        int r = i >> 4, c = i & 15;
        As32[r * 18 + c] = Ag32[(m0 + r) * (DIM / 2) + kb32 + c];
      }
    }
    for (int i = tid; i < 128 * 32; i += 256) {
      int r = i >> 5, c = i & 31;
      Bs[r * 36 + c] = (_Float16)Wout[(n0 + r) * DIM + kb + c];
    }
    __syncthreads();
    v16h a0 = frag_a(As, wm * 32, 0, 36);
    v16h a1 = frag_a(As, wm * 32 + 16, 0, 36);
#pragma unroll
    for (int tn = 0; tn < 4; ++tn) {
      v16h b = frag_b(Bs, wn * 64 + tn * 16, 0, 36);
      acc[0][tn] = wmma_f16(a0, b, acc[0][tn]);
      acc[1][tn] = wmma_f16(a1, b, acc[1][tn]);
    }
  }

#pragma unroll
  for (int tm = 0; tm < 2; ++tm) {
#pragma unroll
    for (int tn = 0; tn < 4; ++tn) {
      int n = n0 + wn * 64 + tn * 16 + (lane & 15);
#pragma unroll
      for (int r = 0; r < 8; ++r) {
        int m = m0 + wm * 32 + tm * 16 + r + 8 * (lane >> 4);
        out[(size_t)m * DIM + n] = acc[tm][tn][r];
      }
    }
  }
}

// ---------------------------------------------------------------------------
extern "C" void kernel_launch(void* const* d_in, const int* in_sizes, int n_in,
                              void* d_out, int out_size, void* d_ws,
                              size_t ws_size, hipStream_t stream) {
  (void)in_sizes; (void)n_in; (void)out_size; (void)ws_size;
  const float* x    = (const float*)d_in[0];
  const float* Wqkv = (const float*)d_in[1];
  const float* Wout = (const float*)d_in[2];
  float* out = (float*)d_out;

  const size_t per = (size_t)HEADS * BATCH * SEQ * DK;  // 8,388,608 halfs
  _Float16* Qb = (_Float16*)d_ws;
  _Float16* Kb = Qb + per;
  _Float16* Vb = Kb + per;
  _Float16* AO = Vb + per;  // total ws use: 4*per*2 = 64 MB

  qkv_rope_kernel<<<dim3(24, 64), 256, 0, stream>>>(x, Wqkv, Qb, Kb, Vb);
  attn_kernel<<<dim3(SEQ / 64, HEADS * BATCH), 128, 0, stream>>>(Qb, Kb, Vb, AO);
  outproj_kernel<<<dim3(8, 64), 256, 0, stream>>>(AO, Wout, out);
}
